// TrajPredAttNet_54296976556452
// MI455X (gfx1250) — matrix operand order repaired and checked by reading, not compiled, not run on hardware
//
#include <hip/hip_runtime.h>
#include <hip/hip_bf16.h>
#include <math.h>

#define NACT 10000
#define NNODE 50000
#define EACT 160000
#define ENODE 250000
#define ENA 160000
#define DIM 128
#define HC 256          // heads * C = 2 * 128
#define EDGEDIM 34
#define EDGEPAD 64      // K padded to multiple of 32
#define NPRED2 60       // 2 * NUM_PREDS
#define NPREDPAD 64     // N padded to multiple of 16

typedef __bf16 bf16;
typedef __attribute__((ext_vector_type(16))) __bf16 v16bf;
typedef __attribute__((ext_vector_type(8)))  float  v8f;

__device__ __forceinline__ bf16 f2bf(float f) {
    unsigned u = __float_as_uint(f);
    unsigned r = u + 0x7fffu + ((u >> 16) & 1u);   // round-to-nearest-even
    unsigned short h = (unsigned short)(r >> 16);
    bf16 b;
    __builtin_memcpy(&b, &h, 2);
    return b;
}

__device__ __forceinline__ void atomicMaxF(float* addr, float val) {
    if (val >= 0.0f) atomicMax((int*)addr, __float_as_int(val));
    else             atomicMin((unsigned int*)addr, __float_as_uint(val));
}

// float -> bf16 weight conversion (weights are tiny; done once per GEMM)
__global__ void k_cvt(const float* __restrict__ src, bf16* __restrict__ dst, int n) {
    int i = blockIdx.x * blockDim.x + threadIdx.x;
    if (i < n) dst[i] = f2bf(src[i]);
}

// ---------------------------------------------------------------------------
// OUT[M,N] = A[M,K] @ W[N,K]^T + bias   (bf16 WMMA, fp32 accumulate)
// Preconditions: M%16==0, N%64==0, K%32==0 -> no predication anywhere.
// One wave computes a 16x64 tile: 4 accumulators share one A fragment,
// B operand is pre-converted bf16 (16-bit loads straight into half-regs).
// Fragments per CDNA5 ISA 7.12.2 layouts.
// ---------------------------------------------------------------------------
template <int K>
__global__ void k_linear_t(const float* __restrict__ A, const bf16* __restrict__ Wb,
                           const float* __restrict__ bias, float* __restrict__ OUT,
                           int M, int N, int relu) {
    int tilesN = N >> 6;
    int wave = blockIdx.x * (blockDim.x >> 5) + (threadIdx.x >> 5);
    if (wave >= (M >> 4) * tilesN) return;          // wave-uniform exit
    int tm = wave / tilesN, tn = wave % tilesN;
    int m0 = tm << 4, n0 = tn << 6;
    int lane = threadIdx.x & 31;
    int half = lane >> 4;
    int r = lane & 15;

    const float* Arow = A + (size_t)(m0 + r) * K + half * 8;
    const bf16* Wcol = Wb + (size_t)(n0 + r) * K + half * 8;  // b_t[j] = Wcol[t*16*K + kb + j + (j>=8 ? 8 : 0)]
    __builtin_prefetch(Arow, 0, 0);

    v8f acc0 = {}, acc1 = {}, acc2 = {}, acc3 = {};
    #pragma unroll
    for (int kb = 0; kb < K; kb += 32) {
        // A 16x32 bf16 fragment: lane<16 -> K {0..7,16..23}; lane>=16 -> {8..15,24..31}
        float4 a0 = *(const float4*)(Arow + kb);
        float4 a1 = *(const float4*)(Arow + kb + 4);
        float4 a2 = *(const float4*)(Arow + kb + 16);
        float4 a3 = *(const float4*)(Arow + kb + 20);
        v16bf a;
        a[0] = f2bf(a0.x);  a[1] = f2bf(a0.y);  a[2]  = f2bf(a0.z);  a[3]  = f2bf(a0.w);
        a[4] = f2bf(a1.x);  a[5] = f2bf(a1.y);  a[6]  = f2bf(a1.z);  a[7]  = f2bf(a1.w);
        a[8] = f2bf(a2.x);  a[9] = f2bf(a2.y);  a[10] = f2bf(a2.z);  a[11] = f2bf(a2.w);
        a[12] = f2bf(a3.x); a[13] = f2bf(a3.y); a[14] = f2bf(a3.z); a[15] = f2bf(a3.w);
        // B 32x16 bf16 fragments: lane holds column n0+t*16+(lane&15); same K order as A
        v16bf b0, b1, b2, b3;
        #pragma unroll
        for (int j = 0; j < 16; ++j) {
            int kk = kb + j + (j >= 8 ? 8 : 0);
            b0[j] = Wcol[(size_t)0  * K + kk];
            b1[j] = Wcol[(size_t)16 * K + kk];
            b2[j] = Wcol[(size_t)32 * K + kk];
            b3[j] = Wcol[(size_t)48 * K + kk];
        }
        acc0 = __builtin_amdgcn_wmma_f32_16x16x32_bf16(false, a, false, b0, (short)0, acc0, false, false);
        acc1 = __builtin_amdgcn_wmma_f32_16x16x32_bf16(false, a, false, b1, (short)0, acc1, false, false);
        acc2 = __builtin_amdgcn_wmma_f32_16x16x32_bf16(false, a, false, b2, (short)0, acc2, false, false);
        acc3 = __builtin_amdgcn_wmma_f32_16x16x32_bf16(false, a, false, b3, (short)0, acc3, false, false);
    }
    // C/D layout: lane -> N = (lane&15); VGPR vr -> M = m0 + vr + 8*half
    v8f accs[4] = {acc0, acc1, acc2, acc3};
    #pragma unroll
    for (int t = 0; t < 4; ++t) {
        int col = n0 + t * 16 + r;
        float bb = bias ? bias[col] : 0.0f;
        #pragma unroll
        for (int vr = 0; vr < 8; ++vr) {
            float v = accs[t][vr] + bb;
            if (relu) v = fmaxf(v, 0.0f);
            OUT[(size_t)(m0 + vr + 8 * half) * N + col] = v;
        }
    }
}

// ---------------------------------------------------------------------------
// per-row LayerNorm over DIM=128 (one wave per row), optional residual & relu
// ---------------------------------------------------------------------------
__global__ void k_ln(const float* __restrict__ X, const float* __restrict__ Res,
                     const float* __restrict__ gs, const float* __restrict__ gb,
                     float* __restrict__ OUT, int rows, int relu) {
    int row = blockIdx.x * (blockDim.x >> 5) + (threadIdx.x >> 5);
    if (row >= rows) return;
    int lane = threadIdx.x & 31;
    float4 xv = ((const float4*)(X + (size_t)row * DIM))[lane];
    if (Res) {
        float4 rv = ((const float4*)(Res + (size_t)row * DIM))[lane];
        xv.x += rv.x; xv.y += rv.y; xv.z += rv.z; xv.w += rv.w;
    }
    float s = xv.x + xv.y + xv.z + xv.w;
    #pragma unroll
    for (int o = 16; o; o >>= 1) s += __shfl_xor(s, o, 32);
    float mean = s * (1.0f / DIM);
    float dx = xv.x - mean, dy = xv.y - mean, dz = xv.z - mean, dw = xv.w - mean;
    float v2 = dx * dx + dy * dy + dz * dz + dw * dw;
    #pragma unroll
    for (int o = 16; o; o >>= 1) v2 += __shfl_xor(v2, o, 32);
    float inv = rsqrtf(v2 * (1.0f / DIM) + 1e-5f);
    int c = lane * 4;
    float o0 = dx * inv * gs[c + 0] + gb[c + 0];
    float o1 = dy * inv * gs[c + 1] + gb[c + 1];
    float o2 = dz * inv * gs[c + 2] + gb[c + 2];
    float o3 = dw * inv * gs[c + 3] + gb[c + 3];
    if (relu) {
        o0 = fmaxf(o0, 0.0f); o1 = fmaxf(o1, 0.0f);
        o2 = fmaxf(o2, 0.0f); o3 = fmaxf(o3, 0.0f);
    }
    float4 ov = {o0, o1, o2, o3};
    ((float4*)(OUT + (size_t)row * DIM))[lane] = ov;
}

__global__ void k_fill(float* __restrict__ p, float v, size_t n) {
    size_t i = blockIdx.x * (size_t)blockDim.x + threadIdx.x;
    if (i < n) p[i] = v;
}

__global__ void k_add_relu(const float* __restrict__ A, const float* __restrict__ B,
                           float* __restrict__ O, size_t n) {
    size_t i = blockIdx.x * (size_t)blockDim.x + threadIdx.x;
    if (i < n) O[i] = fmaxf(A[i] + B[i], 0.0f);
}

// concat [actors | Z_act] -> [NACT, 192]
__global__ void k_concat(const float* __restrict__ act, const float* __restrict__ za,
                         float* __restrict__ O) {
    int i = blockIdx.x * blockDim.x + threadIdx.x;
    if (i >= NACT * 192) return;
    int row = i / 192, col = i - row * 192;
    O[i] = (col < DIM) ? act[(size_t)row * DIM + col]
                       : za[(size_t)row * 64 + (col - DIM)];
}

// dists[e] = concat(Z_int[e], last_pos[src]-last_pos[dst], zeros) -> [EACT, 64]
__global__ void k_dists(const float* __restrict__ zi, const float* __restrict__ lp,
                        const int* __restrict__ ei, float* __restrict__ O) {
    int i = blockIdx.x * blockDim.x + threadIdx.x;
    if (i >= EACT * EDGEPAD) return;
    int e = i >> 6, j = i & 63;
    float v = 0.0f;
    if (j < 32) v = zi[(size_t)e * 32 + j];
    else if (j < EDGEDIM) {
        int s = ei[e], d = ei[EACT + e];
        int c = j - 32;
        v = lp[(size_t)s * 2 + c] - lp[(size_t)d * 2 + c];
    }
    O[i] = v;
}

// zero-pad columns: dst[rows, cols_dst] <- src[rows, cols_src]
__global__ void k_pad_cols(const float* __restrict__ src, float* __restrict__ dst,
                           int rows, int cols_src, int cols_dst) {
    int i = blockIdx.x * blockDim.x + threadIdx.x;
    if (i >= rows * cols_dst) return;
    int r = i / cols_dst, c = i - r * cols_dst;
    dst[i] = (c < cols_src) ? src[(size_t)r * cols_src + c] : 0.0f;
}

// zero-pad rows (same column stride)
__global__ void k_pad_rows(const float* __restrict__ src, float* __restrict__ dst,
                           int rows_src, int rows_dst, int cols) {
    int i = blockIdx.x * blockDim.x + threadIdx.x;
    if (i >= rows_dst * cols) return;
    int r = i / cols;
    dst[i] = (r < rows_src) ? src[i] : 0.0f;
}

// compact columns: dst[rows, cols_dst] <- src[rows, cols_src] (cols_dst < cols_src)
__global__ void k_compact(const float* __restrict__ src, float* __restrict__ dst,
                          int rows, int cols_src, int cols_dst) {
    int i = blockIdx.x * blockDim.x + threadIdx.x;
    if (i >= rows * cols_dst) return;
    int r = i / cols_dst, c = i - r * cols_dst;
    dst[i] = src[(size_t)r * cols_src + c];
}

// logits[e,h] = dot(Q[dst,h], K[src,h] (+E[e,h])) * scale ; atomic max per (dst,h)
__global__ void k_logits(const float* __restrict__ Q, const float* __restrict__ Kb,
                         const float* __restrict__ Ee, const int* __restrict__ ei,
                         int E, float* __restrict__ logits, float* __restrict__ mx,
                         float scale) {
    int idx = blockIdx.x * blockDim.x + threadIdx.x;
    if (idx >= E * 2) return;
    int e = idx >> 1, h = idx & 1;
    int s = ei[e], d = ei[E + e];
    const float4* q4 = (const float4*)(Q + (size_t)d * HC + h * DIM);
    const float4* k4 = (const float4*)(Kb + (size_t)s * HC + h * DIM);
    float acc = 0.0f;
    if (Ee) {
        const float4* e4 = (const float4*)(Ee + (size_t)e * HC + h * DIM);
        for (int c = 0; c < 32; ++c) {
            float4 qv = q4[c], kv = k4[c], ev = e4[c];
            acc += qv.x * (kv.x + ev.x) + qv.y * (kv.y + ev.y)
                 + qv.z * (kv.z + ev.z) + qv.w * (kv.w + ev.w);
        }
    } else {
        for (int c = 0; c < 32; ++c) {
            float4 qv = q4[c], kv = k4[c];
            acc += qv.x * kv.x + qv.y * kv.y + qv.z * kv.z + qv.w * kv.w;
        }
    }
    float l = acc * scale;
    logits[idx] = l;
    atomicMaxF(&mx[(size_t)d * 2 + h], l);
}

// ea = exp(l - mx[dst]); den[dst] += ea  (logits buffer overwritten with ea)
__global__ void k_expsum(float* __restrict__ logits, const float* __restrict__ mx,
                         float* __restrict__ den, const int* __restrict__ ei, int E) {
    int idx = blockIdx.x * blockDim.x + threadIdx.x;
    if (idx >= E * 2) return;
    int e = idx >> 1, h = idx & 1;
    int d = ei[E + e];
    float ea = __expf(logits[idx] - mx[(size_t)d * 2 + h]);
    logits[idx] = ea;
    atomicAdd(&den[(size_t)d * 2 + h], ea);
}

// agg[dst,j] += (V[src,j] (+E[e,j])) * ea[e,h]/den[dst,h]
__global__ void k_agg(const float* __restrict__ Vb, const float* __restrict__ Ee,
                      const float* __restrict__ ea, const float* __restrict__ den,
                      const int* __restrict__ ei, int E, float* __restrict__ agg) {
    size_t idx = blockIdx.x * (size_t)blockDim.x + threadIdx.x;
    if (idx >= (size_t)E * HC) return;
    int e = (int)(idx >> 8);
    int j = (int)(idx & 255);
    int h = j >> 7;
    int s = ei[e], d = ei[E + e];
    float w = ea[(size_t)e * 2 + h] / den[(size_t)d * 2 + h];
    float val = Vb[(size_t)s * HC + j];
    if (Ee) val += Ee[(size_t)e * HC + j];
    atomicAdd(&agg[(size_t)d * HC + j], val * w);
}

// ---------------------------------------------------------------------------
// host-side orchestration
// ---------------------------------------------------------------------------
struct TParams {
    const float *Wq, *bq, *Wk, *bk, *Wv, *bv, *Ws, *bs, *n1s, *n1b;
    const float *We;                                  // nullable
    const float *W1, *b1, *W2, *b2, *n2s, *n2b;       // valid iff root
    bool root;
};

struct Scratch {
    float *Q, *Kb, *Vb, *Eemb, *agg, *linb, *o1, *f1, *f2, *logits, *mx, *den, *We64;
    bf16* wbf;                                        // bf16 weight staging
};

static inline void launch_linear(hipStream_t st, const float* A, const float* W,
                                 const float* b, float* OUT, int M, int N, int K,
                                 int relu, bf16* wbf) {
    k_cvt<<<(N * K + 255) / 256, 256, 0, st>>>(W, wbf, N * K);
    int tiles = (M >> 4) * (N >> 6);
    int blocks = (tiles + 3) / 4;
    switch (K) {
    case 64:  k_linear_t<64><<<blocks, 128, 0, st>>>(A, wbf, b, OUT, M, N, relu); break;
    case 128: k_linear_t<128><<<blocks, 128, 0, st>>>(A, wbf, b, OUT, M, N, relu); break;
    case 192: k_linear_t<192><<<blocks, 128, 0, st>>>(A, wbf, b, OUT, M, N, relu); break;
    case 256: k_linear_t<256><<<blocks, 128, 0, st>>>(A, wbf, b, OUT, M, N, relu); break;
    default: break;
    }
}

static inline void launch_ln(hipStream_t st, const float* X, const float* Res,
                             const float* gs, const float* gb, float* O, int rows,
                             int relu) {
    k_ln<<<(rows + 7) / 8, 256, 0, st>>>(X, Res, gs, gb, O, rows, relu);
}

static void run_tconv(hipStream_t st, const float* xsrc, int Ns, const float* xdst,
                      int Nd, const int* ei, int E, const float* edge_attr,
                      const TParams& P, float* outBuf, const Scratch& S) {
    launch_linear(st, xdst, P.Wq, P.bq, S.Q, Nd, HC, DIM, 0, S.wbf);
    launch_linear(st, xsrc, P.Wk, P.bk, S.Kb, Ns, HC, DIM, 0, S.wbf);
    launch_linear(st, xsrc, P.Wv, P.bv, S.Vb, Ns, HC, DIM, 0, S.wbf);
    const float* Ee = nullptr;
    if (P.We) {
        k_pad_cols<<<(HC * EDGEPAD + 255) / 256, 256, 0, st>>>(P.We, S.We64, HC,
                                                               EDGEDIM, EDGEPAD);
        launch_linear(st, edge_attr, S.We64, nullptr, S.Eemb, E, HC, EDGEPAD, 0, S.wbf);
        Ee = S.Eemb;
    }
    size_t n2 = (size_t)Nd * 2, nagg = (size_t)Nd * HC;
    k_fill<<<(int)((n2 + 255) / 256), 256, 0, st>>>(S.mx, -INFINITY, n2);
    k_fill<<<(int)((n2 + 255) / 256), 256, 0, st>>>(S.den, 0.0f, n2);
    k_fill<<<(int)((nagg + 255) / 256), 256, 0, st>>>(S.agg, 0.0f, nagg);
    const float scale = 0.08838834764831845f;         // 1/sqrt(128)
    k_logits<<<(E * 2 + 255) / 256, 256, 0, st>>>(S.Q, S.Kb, Ee, ei, E, S.logits,
                                                  S.mx, scale);
    k_expsum<<<(E * 2 + 255) / 256, 256, 0, st>>>(S.logits, S.mx, S.den, ei, E);
    k_agg<<<E, 256, 0, st>>>(S.Vb, Ee, S.logits, S.den, ei, E, S.agg);
    launch_linear(st, S.agg, P.Ws, P.bs, S.linb, Nd, DIM, HC, 0, S.wbf);
    if (P.root) {
        launch_ln(st, S.linb, xdst, P.n1s, P.n1b, S.o1, Nd, 0);
        launch_linear(st, S.o1, P.W1, P.b1, S.f1, Nd, HC, DIM, 1, S.wbf);
        launch_linear(st, S.f1, P.W2, P.b2, S.f2, Nd, DIM, HC, 0, S.wbf);
        launch_ln(st, S.f2, xdst, P.n2s, P.n2b, outBuf, Nd, 0);
    } else {
        launch_ln(st, S.linb, xdst, P.n1s, P.n1b, outBuf, Nd, 0);
    }
}

extern "C" void kernel_launch(void* const* d_in, const int* in_sizes, int n_in,
                              void* d_out, int out_size, void* d_ws, size_t ws_size,
                              hipStream_t stream) {
    (void)in_sizes; (void)n_in; (void)out_size; (void)ws_size;
    auto F = [&](int i) -> const float* { return (const float*)d_in[i]; };
    auto I = [&](int i) -> const int*   { return (const int*)d_in[i]; };

    const float* actors   = F(0);
    const float* last_pos = F(2);
    const float* Z_act    = F(4);
    const float* Z_int    = F(5);
    const int* aei  = I(6);
    const int* nei  = I(7);
    const int* naei = I(8);
    const float* embed_W = F(9);
    const float* embed_b = F(10);
    const int PATH_BASE = 11, INT_BASE = 43, TRAJ_BASE = 65;
    const float* lr1_W = F(97); const float* gn1_s = F(98); const float* gn1_b = F(99);
    const float* lr2_W = F(100); const float* gn2_s = F(101); const float* gn2_b = F(102);
    const float* out_W = F(103); const float* out_b = F(104);

    auto blk16 = [&](int base) {
        TParams p{};
        p.Wq = F(base); p.bq = F(base + 1); p.Wk = F(base + 2); p.bk = F(base + 3);
        p.Wv = F(base + 4); p.bv = F(base + 5); p.Ws = F(base + 6); p.bs = F(base + 7);
        p.n1s = F(base + 8); p.n1b = F(base + 9);
        p.W1 = F(base + 10); p.b1 = F(base + 11); p.W2 = F(base + 12); p.b2 = F(base + 13);
        p.n2s = F(base + 14); p.n2b = F(base + 15);
        p.We = nullptr; p.root = true;
        return p;
    };
    auto blk11 = [&](int base) {
        TParams p{};
        p.Wq = F(base); p.bq = F(base + 1); p.Wk = F(base + 2); p.bk = F(base + 3);
        p.Wv = F(base + 4); p.bv = F(base + 5); p.Ws = F(base + 6); p.bs = F(base + 7);
        p.n1s = F(base + 8); p.n1b = F(base + 9);
        p.We = F(base + 10); p.root = false;
        return p;
    };

    // ---- workspace arena ----
    char* base = (char*)d_ws;
    size_t off = 0;
    auto alloc = [&](size_t nfloat) -> float* {
        float* p = (float*)(base + off);
        off = (off + nfloat * sizeof(float) + 255) & ~(size_t)255;
        return p;
    };
    float* Acat    = alloc((size_t)NACT * 192);
    float* nodesA  = alloc((size_t)NNODE * DIM);
    float* nodesB  = alloc((size_t)NNODE * DIM);
    float* actA    = alloc((size_t)NACT * DIM);
    float* actB    = alloc((size_t)NACT * DIM);
    float* dists   = alloc((size_t)EACT * EDGEPAD);
    float* outW64  = alloc((size_t)NPREDPAD * DIM);
    float* outb64  = alloc((size_t)NPREDPAD);
    float* headtmp = alloc((size_t)NACT * NPREDPAD);
    Scratch S;
    S.Q      = alloc((size_t)NNODE * HC);
    S.Kb     = alloc((size_t)NNODE * HC);
    S.Vb     = alloc((size_t)NNODE * HC);
    S.Eemb   = alloc((size_t)EACT * HC);
    S.agg    = alloc((size_t)NNODE * HC);
    S.linb   = alloc((size_t)NNODE * DIM);
    S.o1     = alloc((size_t)NNODE * DIM);
    S.f1     = alloc((size_t)NNODE * HC);
    S.f2     = alloc((size_t)NNODE * DIM);
    S.logits = alloc((size_t)ENODE * 2);
    S.mx     = alloc((size_t)NNODE * 2);
    S.den    = alloc((size_t)NNODE * 2);
    S.We64   = alloc((size_t)HC * EDGEPAD);
    S.wbf    = (bf16*)alloc(32768);                   // 64K bf16 = max weight size

    // ---- stage 0: embed actors ----
    k_concat<<<(NACT * 192 + 255) / 256, 256, 0, stream>>>(actors, Z_act, Acat);
    launch_linear(stream, Acat, embed_W, embed_b, actA, NACT, DIM, 192, 0, S.wbf);

    // ---- stage 1: path blocks on nodes ----
    hipMemcpyAsync(nodesA, d_in[1], (size_t)NNODE * DIM * sizeof(float),
                   hipMemcpyDeviceToDevice, stream);
    float* ncur = nodesA; float* nnxt = nodesB;
    for (int b = 0; b < 2; ++b) {
        TParams P = blk16(PATH_BASE + b * 16);
        run_tconv(stream, ncur, NNODE, ncur, NNODE, nei, ENODE, nullptr, P, nnxt, S);
        float* t = ncur; ncur = nnxt; nnxt = t;
    }

    // ---- stage 2: edge attributes (K zero-padded to 64) ----
    k_dists<<<(EACT * EDGEPAD + 255) / 256, 256, 0, stream>>>(Z_int, last_pos, aei, dists);

    // ---- stage 3: interleaved int / traj blocks on actors ----
    float* acur = actA; float* anxt = actB;
    for (int i = 0; i < 2; ++i) {
        TParams Pi = blk11(INT_BASE + i * 11);
        run_tconv(stream, acur, NACT, acur, NACT, aei, EACT, dists, Pi, anxt, S);
        { float* t = acur; acur = anxt; anxt = t; }
        TParams Pt = blk16(TRAJ_BASE + i * 16);
        run_tconv(stream, ncur, NNODE, acur, NACT, naei, ENA, nullptr, Pt, anxt, S);
        { float* t = acur; acur = anxt; anxt = t; }
    }

    // ---- stage 4: prediction head ----
    float* t1 = S.linb;          // acur @ lr1^T
    float* h1 = S.o1;            // relu(LN(t1))
    float* t2 = S.f2;            // h1 @ lr2^T
    float* h2 = S.linb;          // LN(t2)
    float* h3 = S.f1;            // relu(h2 + acur)
    launch_linear(stream, acur, lr1_W, nullptr, t1, NACT, DIM, DIM, 0, S.wbf);
    launch_ln(stream, t1, nullptr, gn1_s, gn1_b, h1, NACT, 1);
    launch_linear(stream, h1, lr2_W, nullptr, t2, NACT, DIM, DIM, 0, S.wbf);
    launch_ln(stream, t2, nullptr, gn2_s, gn2_b, h2, NACT, 0);
    k_add_relu<<<(NACT * DIM + 255) / 256, 256, 0, stream>>>(h2, acur, h3,
                                                             (size_t)NACT * DIM);
    // pad out_W/out_b to N=64, GEMM, then compact 60 columns into d_out
    k_pad_rows<<<(NPREDPAD * DIM + 255) / 256, 256, 0, stream>>>(out_W, outW64,
                                                                 NPRED2, NPREDPAD, DIM);
    k_pad_rows<<<1, 64, 0, stream>>>(out_b, outb64, NPRED2, NPREDPAD, 1);
    launch_linear(stream, h3, outW64, outb64, headtmp, NACT, NPREDPAD, DIM, 0, S.wbf);
    k_compact<<<(NACT * NPRED2 + 255) / 256, 256, 0, stream>>>(headtmp, (float*)d_out,
                                                               NACT, NPREDPAD, NPRED2);
}
